// LinkPredictor_66571993088846
// MI455X (gfx1250) — matrix operand order repaired and checked by reading, not compiled, not run on hardware
//
#include <hip/hip_runtime.h>
#include <hip/hip_bf16.h>

typedef __attribute__((ext_vector_type(16))) _Float16 v16h;
typedef __attribute__((ext_vector_type(8)))  float    v8f;
typedef __attribute__((ext_vector_type(4)))  int      v4i;
typedef __attribute__((ext_vector_type(4)))  float    v4f;

#define N_NODES  100000
#define IN_CH    128
#define HIDDEN   64
#define N_EDGES  1000000
#define LDA      264                // padded row stride in halves (256+8): conflict-free ds_load_b128
#define WPB      4                  // waves per block (LDS: 33KB W1^T + 4*16.5KB A = ~101KB -> 3 blocks/WGP)
#define ET       2                  // 16-edge tiles per wave iteration (M = 32 edges)
#define N_GROUPS (N_EDGES / (ET * 16))   // 31250

union ABFrag { v16h v; v4i q[2]; };
union H4     { _Float16 h[4]; unsigned long long u; };

__device__ __forceinline__ float sigmoidf(float x) {
    return 1.0f / (1.0f + __expf(-x));
}

// one-shot f32 -> f16 conversion of z into workspace
__global__ __launch_bounds__(256)
void cvt_z_kernel(const float* __restrict__ z, _Float16* __restrict__ zh) {
    int i = blockIdx.x * blockDim.x + threadIdx.x;   // one float4 per thread
    if (i < (N_NODES * IN_CH) / 4) {
        v4f v = *(const v4f*)(z + 4 * i);
        H4 p;
        p.h[0] = (_Float16)v[0];
        p.h[1] = (_Float16)v[1];
        p.h[2] = (_Float16)v[2];
        p.h[3] = (_Float16)v[3];
        *(unsigned long long*)(zh + 4 * i) = p.u;
    }
}

template<bool PRECONV>
__global__ __launch_bounds__(WPB * 32)
void linkpred_wmma_kernel(const float*    __restrict__ z,    // f32 features (fallback path)
                          const _Float16* __restrict__ zh,   // f16 features (fast path)
                          const int*      __restrict__ ep,   // (2, N_EDGES)
                          const float*    __restrict__ w1,   // (256, 64)
                          const float*    __restrict__ b1,   // (64,)
                          const float*    __restrict__ w2,   // (64, 1)
                          const float*    __restrict__ b2,   // (1,)
                          float*          __restrict__ out)  // (N_EDGES,)
{
    __shared__ _Float16 sW[HIDDEN * LDA];             // W1^T as f16: [n][k]
    __shared__ _Float16 sA[WPB][ET * 16 * LDA];       // per-wave A tiles: rows = 32 edges

    const int tid  = threadIdx.x;
    const int wid  = tid >> 5;
    const int lane = tid & 31;
    const int m    = lane & 15;
    const int half = lane >> 4;

    // ---- stage W1^T into LDS as f16 (block cooperative, once) ----
    for (int i = tid; i < (2 * IN_CH * HIDDEN) / 4; i += blockDim.x) {
        int k  = i >> 4;
        int n0 = (i & 15) << 2;
        v4f v = *(const v4f*)(w1 + k * HIDDEN + n0);
        sW[(n0 + 0) * LDA + k] = (_Float16)v[0];
        sW[(n0 + 1) * LDA + k] = (_Float16)v[1];
        sW[(n0 + 2) * LDA + k] = (_Float16)v[2];
        sW[(n0 + 3) * LDA + k] = (_Float16)v[3];
    }
    __syncthreads();

    float b1v[4], w2v[4];
#pragma unroll
    for (int t = 0; t < 4; ++t) {
        b1v[t] = b1[t * 16 + m];
        w2v[t] = w2[t * 16 + m];
    }
    const float b2v = b2[0];

    _Float16* At = &sA[wid][0];

    const int gwave  = blockIdx.x * WPB + wid;
    const int nwaves = gridDim.x * WPB;

    for (int grp = gwave; grp < N_GROUPS; grp += nwaves) {
        const int ebase = grp * (ET * 16);

        // ================= gather 64 node rows into the wave's A tiles =================
        if (PRECONV) {
            // lane owns src/dst endpoint of edge (ebase + lane)
            int idx_s = ep[ebase + lane];
            int idx_d = ep[N_EDGES + ebase + lane];
            // two rows per step: lanes 0-15 row 2s, lanes 16-31 row 2s+1;
            // each lane moves 16B (8 halves) at column 8*m
#pragma unroll 4
            for (int s = 0; s < 16; ++s) {
                int j  = 2 * s + half;                 // edge row 0..31
                int nd = __shfl(idx_s, j, 32);
                v4i v  = *(const v4i*)(zh + (long long)nd * IN_CH + 8 * m);
                *(v4i*)(At + j * LDA + 8 * m) = v;     // src features -> cols 0..127
            }
#pragma unroll 4
            for (int s = 0; s < 16; ++s) {
                int j  = 2 * s + half;
                int nd = __shfl(idx_d, j, 32);
                v4i v  = *(const v4i*)(zh + (long long)nd * IN_CH + 8 * m);
                *(v4i*)(At + j * LDA + IN_CH + 8 * m) = v;  // dst -> cols 128..255
            }
        } else {
            // fallback: gather f32 and convert in-flight
#pragma unroll
            for (int tt = 0; tt < ET; ++tt) {
                int node = ep[half * N_EDGES + ebase + tt * 16 + m];
#pragma unroll 4
                for (int r = 0; r < 32; ++r) {
                    int nd = __shfl(node, r, 32);
                    v4f v  = *(const v4f*)(z + (long long)nd * IN_CH + lane * 4);
                    H4 p;
                    p.h[0] = (_Float16)v[0];
                    p.h[1] = (_Float16)v[1];
                    p.h[2] = (_Float16)v[2];
                    p.h[3] = (_Float16)v[3];
                    int row  = tt * 16 + (r & 15);
                    int koff = (r < 16) ? 0 : IN_CH;
                    *(unsigned long long*)(At + row * LDA + koff + lane * 4) = p.u;
                }
            }
        }

        // wave-private tile: wait on our own ds stores (no block barrier needed)
        asm volatile("s_wait_dscnt 0x0" ::: "memory");

        // ================= (32 x 256) @ (256 x 64): B fragments reused across both tiles =================
        v8f acc[ET][4];
#pragma unroll
        for (int tt = 0; tt < ET; ++tt)
#pragma unroll
            for (int t = 0; t < 4; ++t) acc[tt][t] = (v8f){};

#pragma unroll
        for (int ks = 0; ks < 8; ++ks) {
            const int k0 = ks * 32;
            const int ak = k0 + (half ? 8 : 0);
            ABFrag a[ET];
#pragma unroll
            for (int tt = 0; tt < ET; ++tt) {
                const _Float16* ap = At + (tt * 16 + m) * LDA + ak;
                a[tt].q[0] = *(const v4i*)(ap);
                a[tt].q[1] = *(const v4i*)(ap + 16);
            }
            const int bk = k0 + (half ? 8 : 0);
#pragma unroll
            for (int t = 0; t < 4; ++t) {
                const _Float16* bp = sW + (t * 16 + m) * LDA + bk;
                ABFrag b;
                b.q[0] = *(const v4i*)(bp);
                b.q[1] = *(const v4i*)(bp + 16);
#pragma unroll
                for (int tt = 0; tt < ET; ++tt) {
                    acc[tt][t] = __builtin_amdgcn_wmma_f32_16x16x32_f16(
                        false, a[tt].v, false, b.v, (short)0, acc[tt][t], false, false);
                }
            }
        }

        // ================= epilogue per tile: relu+bias, dot w2, reduce, sigmoid =================
#pragma unroll
        for (int tt = 0; tt < ET; ++tt) {
            float s[8];
#pragma unroll
            for (int j = 0; j < 8; ++j) s[j] = 0.0f;
#pragma unroll
            for (int t = 0; t < 4; ++t) {
#pragma unroll
                for (int j = 0; j < 8; ++j) {
                    float h = acc[tt][t][j] + b1v[t];
                    h = fmaxf(h, 0.0f);
                    s[j] = fmaf(h, w2v[t], s[j]);
                }
            }
#pragma unroll
            for (int j = 0; j < 8; ++j) {
                s[j] += __shfl_xor(s[j], 1, 32);
                s[j] += __shfl_xor(s[j], 2, 32);
                s[j] += __shfl_xor(s[j], 4, 32);
                s[j] += __shfl_xor(s[j], 8, 32);
            }
#pragma unroll
            for (int j = 0; j < 8; ++j) {
                if (m == j) {
                    out[ebase + tt * 16 + half * 8 + j] = sigmoidf(s[j] + b2v);
                }
            }
        }
    }
}

extern "C" void kernel_launch(void* const* d_in, const int* in_sizes, int n_in,
                              void* d_out, int out_size, void* d_ws, size_t ws_size,
                              hipStream_t stream) {
    const float* z  = (const float*)d_in[0];
    const int*   ep = (const int*)d_in[1];
    const float* w1 = (const float*)d_in[2];
    const float* b1 = (const float*)d_in[3];
    const float* w2 = (const float*)d_in[4];
    const float* b2 = (const float*)d_in[5];
    float* out = (float*)d_out;

    _Float16* zh = (_Float16*)d_ws;
    const size_t zh_bytes = (size_t)N_NODES * IN_CH * 2;

    dim3 grid(1024);
    dim3 block(WPB * 32);

    if (ws_size >= zh_bytes) {
        cvt_z_kernel<<<dim3((N_NODES * IN_CH / 4 + 255) / 256), dim3(256), 0, stream>>>(z, zh);
        linkpred_wmma_kernel<true><<<grid, block, 0, stream>>>(z, zh, ep, w1, b1, w2, b2, out);
    } else {
        linkpred_wmma_kernel<false><<<grid, block, 0, stream>>>(z, zh, ep, w1, b1, w2, b2, out);
    }
}
